// Conditioner_25005299597437
// MI455X (gfx1250) — matrix operand order repaired and checked by reading, not compiled, not run on hardware
//
#include <hip/hip_runtime.h>
#include <hip/hip_bf16.h>

typedef __attribute__((ext_vector_type(16))) __bf16 v16bf;
typedef __attribute__((ext_vector_type(8)))  __bf16 v8bf;
typedef __attribute__((ext_vector_type(8)))  float  v8f;

#define P_      8
#define T_      2048
#define N_      8192
#define M_      20000
#define DLAT    1024
#define DFEAT   512
#define KTOT    1536
#define ROWS    16384   // P_*T_

#define KT      32      // K per WMMA step
#define RT      128     // rows per block
#define CT      128     // cols per block

// ---------------------------------------------------------------------------
// Kernel 0: one-shot split of W [KTOT, DLAT] f32 into bf16 hi/lo, transposed
// to [DLAT, KTOT] so GEMM B-staging is a contiguous 16B-aligned copy.
// ---------------------------------------------------------------------------
__global__ __launch_bounds__(256) void split_w_kernel(
    const float* __restrict__ W,      // [KTOT, DLAT]
    __bf16* __restrict__ WtHi,        // [DLAT, KTOT]
    __bf16* __restrict__ WtLo)        // [DLAT, KTOT]
{
    const int idx = blockIdx.x * 256 + threadIdx.x;   // coalesced read
    const int k = idx / DLAT;
    const int c = idx - k * DLAT;
    const float v = W[idx];
    const __bf16 h = (__bf16)v;
    WtHi[(size_t)c * KTOT + k] = h;
    WtLo[(size_t)c * KTOT + k] = (__bf16)(v - (float)h);
}

// ---------------------------------------------------------------------------
// Kernel 1: nearest_idx[n] = argmin_m (-2*dot(a_n, b_m) + |b_m|^2)
// a = object_surface[0, n, 0:3]
// ---------------------------------------------------------------------------
__global__ __launch_bounds__(256) void nn_precomp_kernel(
    const float* __restrict__ surf,   // [P, N, 6] (batch 0 used)
    const float* __restrict__ pts,    // [M, 3]
    int* __restrict__ nearest)        // [N]
{
    __shared__ float sx[256], sy[256], sz[256], sq[256];
    const int tid = threadIdx.x;
    const int n = blockIdx.x * 256 + tid;
    const float ax = surf[n * 6 + 0];
    const float ay = surf[n * 6 + 1];
    const float az = surf[n * 6 + 2];
    const float nax = -2.0f * ax, nay = -2.0f * ay, naz = -2.0f * az;

    float best = 3.4e38f;
    int   bi = 0;
    for (int base = 0; base < M_; base += 256) {
        const int cnt = min(256, M_ - base);
        if (tid < cnt) {
            const float px = pts[(base + tid) * 3 + 0];
            const float py = pts[(base + tid) * 3 + 1];
            const float pz = pts[(base + tid) * 3 + 2];
            sx[tid] = px; sy[tid] = py; sz[tid] = pz;
            sq[tid] = px * px + py * py + pz * pz;
        }
        __syncthreads();
        for (int j = 0; j < cnt; ++j) {
            const float d = fmaf(nax, sx[j], fmaf(nay, sy[j], fmaf(naz, sz[j], sq[j])));
            if (d < best) { best = d; bi = base + j; }
        }
        __syncthreads();
    }
    nearest[n] = bi;
}

// ---------------------------------------------------------------------------
// Kernel 2: fused global/local matches -> combined M-indices
// ---------------------------------------------------------------------------
__global__ __launch_bounds__(256) void match_kernel(
    const float* __restrict__ surf,   // [P, N, 6]
    const float* __restrict__ gpc,    // [P, T, 3]
    const float* __restrict__ lpc,    // [P, T, 3]
    const int*   __restrict__ nearest,// [N]
    int* __restrict__ idxMg,          // [P*T]
    int* __restrict__ idxMl)          // [P*T]
{
    __shared__ float sx[256], sy[256], sz[256], sq[256];
    const int tid = threadIdx.x;
    const int p = blockIdx.y;
    const int t = blockIdx.x * 256 + tid;
    const int row = p * T_ + t;

    const float gx = gpc[row * 3 + 0], gy = gpc[row * 3 + 1], gz = gpc[row * 3 + 2];
    const float lx = lpc[row * 3 + 0], ly = lpc[row * 3 + 1], lz = lpc[row * 3 + 2];
    const float* sp = surf + (size_t)p * N_ * 6;

    float bestg = 3.4e38f, bestl = 3.4e38f;
    int ig = 0, il = 0;
    for (int base = 0; base < N_; base += 256) {
        const float bx = sp[(base + tid) * 6 + 0];
        const float by = sp[(base + tid) * 6 + 1];
        const float bz = sp[(base + tid) * 6 + 2];
        sx[tid] = bx; sy[tid] = by; sz[tid] = bz;
        sq[tid] = bx * bx + by * by + bz * bz;
        __syncthreads();
        #pragma unroll 4
        for (int j = 0; j < 256; ++j) {
            const float bxj = sx[j], byj = sy[j], bzj = sz[j], sqj = sq[j];
            const float dg = fmaf(-2.0f * gx, bxj, fmaf(-2.0f * gy, byj, fmaf(-2.0f * gz, bzj, sqj)));
            if (dg < bestg) { bestg = dg; ig = base + j; }
            const float dl = fmaf(-2.0f * lx, bxj, fmaf(-2.0f * ly, byj, fmaf(-2.0f * lz, bzj, sqj)));
            if (dl < bestl) { bestl = dl; il = base + j; }
        }
        __syncthreads();
    }
    idxMg[row] = nearest[ig];
    idxMl[row] = nearest[il];
}

// ---------------------------------------------------------------------------
// Kernel 3: out[row, :] = concat(raw[row,0:1024], feats[idxM[row],0:512]) @ W + bias
// Split-bf16 WMMA (AhiBhi + AhiBlo + AloBhi), f32 accumulate.
// Block = 256 thr (8 waves); tile = 128 rows x 128 cols; K-step = 32.
// Each wave: 16 rows x 128 cols = 8 accumulators, 24 WMMA / K-step.
// B tile staged via async global->LDS (pre-split, pre-transposed bf16 weights).
// ---------------------------------------------------------------------------
__global__ __launch_bounds__(256) void gemm_cond_kernel(
    const float*  __restrict__ raw,    // [ROWS, DLAT]
    const float*  __restrict__ feats,  // [M, DFEAT]
    const int*    __restrict__ idxM,   // [ROWS]
    const __bf16* __restrict__ WtHi,   // [DLAT, KTOT]
    const __bf16* __restrict__ WtLo,   // [DLAT, KTOT]
    const float*  __restrict__ bias,   // [DLAT]
    float* __restrict__ out)           // [ROWS, DLAT]
{
    __shared__ alignas(64) __bf16 Ahi[RT][KT];
    __shared__ alignas(64) __bf16 Alo[RT][KT];
    __shared__ alignas(64) __bf16 Bhi[CT][KT];   // [col][k]
    __shared__ alignas(64) __bf16 Blo[CT][KT];   // [col][k]

    const int tid = threadIdx.x;
    const int rowBase = blockIdx.x * RT;
    const int colBase = blockIdx.y * CT;
    const int wave = tid >> 5;              // 0..7 : 16-row strip per wave
    const int lane = tid & 31;
    const int m  = lane & 15;
    const int kbaseA = (lane >> 4) << 4;    // 0 or 16 (A: K kb..kb+15)
    const int kbaseB = (lane >> 4) << 4;    // 0 or 16 (B: K 0..15 / 16..31)

    // ---- A staging assignment: 16 consecutive k's of one row per thread ----
    const int ar  = tid >> 1;               // 0..127 row in tile
    const int kk0 = (tid & 1) * 16;         // 0 or 16
    const int grow = rowBase + ar;
    const float* __restrict__ rawRow  = raw + (size_t)grow * DLAT;
    const float* __restrict__ featRow = feats + (size_t)idxM[grow] * DFEAT;

    // ---- B staging assignment: async copy 32B (hi) + 32B (lo) per thread ----
    const int bcs = tid >> 1;               // 0..127 col in tile
    const int bh  = (tid & 1) * 16;         // k-half
    const __bf16* gHiBase = WtHi + (size_t)(colBase + bcs) * KTOT + bh;
    const __bf16* gLoBase = WtLo + (size_t)(colBase + bcs) * KTOT + bh;
    const unsigned ldsHi = (unsigned)(uintptr_t)&Bhi[bcs][bh];
    const unsigned ldsLo = (unsigned)(uintptr_t)&Blo[bcs][bh];

    v8f acc[8] = {};
    union V16 { v16bf v; v8bf h[2]; };

    for (int k0 = 0; k0 < KTOT; k0 += KT) {
        // ---- B tile: async global->LDS, 4 x b128 per thread (ASYNCcnt) ----
        {
            const unsigned long long aH = (unsigned long long)(uintptr_t)(gHiBase + k0);
            const unsigned long long aL = (unsigned long long)(uintptr_t)(gLoBase + k0);
            asm volatile("global_load_async_to_lds_b128 %0, %1, off"
                         :: "v"(ldsHi), "v"(aH) : "memory");
            asm volatile("global_load_async_to_lds_b128 %0, %1, off offset:16"
                         :: "v"(ldsHi), "v"(aH) : "memory");
            asm volatile("global_load_async_to_lds_b128 %0, %1, off"
                         :: "v"(ldsLo), "v"(aL) : "memory");
            asm volatile("global_load_async_to_lds_b128 %0, %1, off offset:16"
                         :: "v"(ldsLo), "v"(aL) : "memory");
        }
        // ---- A tile (128x32): 16 f32 per thread, split bf16 hi/lo ----
        {
            const int k = k0 + kk0;
            const float* src = (k < DLAT) ? (rawRow + k) : (featRow + (k - DLAT));
            #pragma unroll
            for (int q = 0; q < 4; ++q) {
                const float4 v = *(const float4*)(src + q * 4);
                const float vv[4] = { v.x, v.y, v.z, v.w };
                #pragma unroll
                for (int j = 0; j < 4; ++j) {
                    const __bf16 hh = (__bf16)vv[j];
                    Ahi[ar][kk0 + q * 4 + j] = hh;
                    Alo[ar][kk0 + q * 4 + j] = (__bf16)(vv[j] - (float)hh);
                }
            }
            if (k0 + KT < KTOT) {
                const int kn = k0 + KT + kk0;
                __builtin_prefetch((kn < DLAT) ? (rawRow + kn) : (featRow + (kn - DLAT)), 0, 1);
            }
        }
        asm volatile("s_wait_asynccnt 0x0" ::: "memory");
        __syncthreads();

        // ---- A fragments (16x32 bf16): two contiguous 16B halves ----
        const int arow = wave * 16 + m;
        V16 aHi, aLo;
        aHi.h[0] = *(const v8bf*)&Ahi[arow][kbaseA];
        aHi.h[1] = *(const v8bf*)&Ahi[arow][kbaseA + 8];
        aLo.h[0] = *(const v8bf*)&Alo[arow][kbaseA];
        aLo.h[1] = *(const v8bf*)&Alo[arow][kbaseA + 8];

        // ---- 8 col-tiles of 16: B fragments + 3 WMMAs each ----
        #pragma unroll
        for (int nt = 0; nt < 8; ++nt) {
            const int bn = nt * 16 + m;
            const v16bf bH = *(const v16bf*)&Bhi[bn][kbaseB];
            const v16bf bL = *(const v16bf*)&Blo[bn][kbaseB];
            acc[nt] = __builtin_amdgcn_wmma_f32_16x16x32_bf16(
                false, aHi.v, false, bH, (short)0, acc[nt], false, false);
            acc[nt] = __builtin_amdgcn_wmma_f32_16x16x32_bf16(
                false, aHi.v, false, bL, (short)0, acc[nt], false, false);
            acc[nt] = __builtin_amdgcn_wmma_f32_16x16x32_bf16(
                false, aLo.v, false, bH, (short)0, acc[nt], false, false);
        }
        __syncthreads();
    }

    // ---- epilogue: bias add + store (D layout: VGPR r -> row r+8*(lane>>4), col lane&15) ----
    const int rOff = (lane >> 4) << 3;
    #pragma unroll
    for (int nt = 0; nt < 8; ++nt) {
        const int col = colBase + nt * 16 + m;
        const float bv = bias[col];
        #pragma unroll
        for (int r = 0; r < 8; ++r) {
            const int row = rowBase + wave * 16 + rOff + r;
            out[(size_t)row * DLAT + col] = acc[nt][r] + bv;
        }
    }
}

// ---------------------------------------------------------------------------
extern "C" void kernel_launch(void* const* d_in, const int* in_sizes, int n_in,
                              void* d_out, int out_size, void* d_ws, size_t ws_size,
                              hipStream_t stream) {
    (void)in_sizes; (void)n_in; (void)out_size; (void)ws_size;

    const float* surf    = (const float*)d_in[0];   // object_surface [8,8192,6]
    const float* pts     = (const float*)d_in[1];   // precomputed_points [20000,3]
    const float* feats   = (const float*)d_in[2];   // precomputed_feats [20000,512]
    const float* geo_raw = (const float*)d_in[3];   // [8,2048,1024]
    const float* obj_raw = (const float*)d_in[4];   // [8,2048,1024]
    const float* lpc     = (const float*)d_in[5];   // local_pc  [8,2048,3]
    const float* gpc     = (const float*)d_in[6];   // global_pc [8,2048,3]
    const float* Wg      = (const float*)d_in[7];   // [1536,1024]
    const float* bg      = (const float*)d_in[8];   // [1024]
    const float* Wo      = (const float*)d_in[9];   // [1536,1024]
    const float* bo      = (const float*)d_in[10];  // [1024]

    float* geo_out = (float*)d_out;                          // [8,2048,1024]
    float* obj_out = (float*)d_out + (size_t)ROWS * DLAT;    // [8,2048,1024]

    // workspace layout
    int* nearest = (int*)d_ws;              // [8192]
    int* idxMg   = nearest + N_;            // [16384]
    int* idxMl   = idxMg + ROWS;            // [16384]
    __bf16* wt   = (__bf16*)(idxMl + ROWS); // 4 x [DLAT*KTOT] bf16 (12 MB)
    const size_t WSZ = (size_t)DLAT * KTOT;
    __bf16* WtHiO = wt;
    __bf16* WtLoO = wt + WSZ;
    __bf16* WtHiG = wt + 2 * WSZ;
    __bf16* WtLoG = wt + 3 * WSZ;

    split_w_kernel<<<(KTOT * DLAT) / 256, 256, 0, stream>>>(Wo, WtHiO, WtLoO);
    split_w_kernel<<<(KTOT * DLAT) / 256, 256, 0, stream>>>(Wg, WtHiG, WtLoG);

    nn_precomp_kernel<<<N_ / 256, 256, 0, stream>>>(surf, pts, nearest);
    match_kernel<<<dim3(T_ / 256, P_), 256, 0, stream>>>(surf, gpc, lpc, nearest, idxMg, idxMl);

    dim3 g3(ROWS / RT, DLAT / CT);
    gemm_cond_kernel<<<g3, 256, 0, stream>>>(obj_raw, feats, idxMg, WtHiO, WtLoO, bo, obj_out);
    gemm_cond_kernel<<<g3, 256, 0, stream>>>(geo_raw, feats, idxMl, WtHiG, WtLoG, bg, geo_out);
}
